// FasterRCNN_44624710206046
// MI455X (gfx1250) — matrix operand order repaired and checked by reading, not compiled, not run on hardware
//
#include <hip/hip_runtime.h>
#include <hip/hip_bf16.h>

// ---------------------------------------------------------------------------
// Problem constants
// ---------------------------------------------------------------------------
#define CCH   1024          // channels
#define FH    100           // feature map H
#define FW    50            // feature map W
#define NROI  128           // proposals
#define KD    50176         // C*7*7 flattened feature length
#define MFC   1024          // fc output dim
#define NSPL  16            // K splits for the big GEMM
#define KCH   (KD / NSPL)   // 3136 K per split (exact)
#define KSTEP 64            // K per staged tile (2 WMMA k-chunks)
#define NIT   (KCH / KSTEP) // 49 iterations per split
#define BSTR  72            // LDS row stride in bf16 elems (16B aligned, conflict-free)
#define NCLS  21

typedef __attribute__((ext_vector_type(16))) __bf16 v16bf;
typedef __attribute__((ext_vector_type(4)))  __bf16 bf4;
typedef __attribute__((ext_vector_type(8)))  float  v8f;
typedef __attribute__((ext_vector_type(4)))  float  f4;

struct __attribute__((aligned(16))) U4 { unsigned int x, y, z, w; };
union Frag {
  U4    q[2];
  v16bf v;
};
union Pack8 {           // 8 bf16 = one 16B LDS store
  bf4 b[2];
  U4  u;
};

__device__ __forceinline__ unsigned short f2bf(float f) {
  unsigned int u = __float_as_uint(f);
  u += 0x7FFFu + ((u >> 16) & 1u);      // round-to-nearest-even
  return (unsigned short)(u >> 16);
}

// ---------------------------------------------------------------------------
// 1) Adaptive 7x7 ROI max pool -> bf16 activations A[NROI][KD]
//    k index = c*49 + py*7 + px  (matches pooled.reshape / fc_w layout)
// ---------------------------------------------------------------------------
__global__ __launch_bounds__(256)
void roi_pool_kernel(const float* __restrict__ fm,
                     const int*   __restrict__ props,
                     unsigned short* __restrict__ A) {
  const int n  = blockIdx.x;
  const int r0 = props[n * 4 + 0] >> 4;
  const int c0 = props[n * 4 + 1] >> 4;
  const int r1 = props[n * 4 + 2] >> 4;
  const int c1 = props[n * 4 + 3] >> 4;
  const int h  = r1 - r0 + 1;
  const int w  = c1 - c0 + 1;

  for (int idx = threadIdx.x; idx < KD; idx += blockDim.x) {
    const int c  = idx / 49;
    const int pp = idx - c * 49;
    const int py = pp / 7;
    const int px = pp - py * 7;
    const int rs = r0 + (py * h) / 7;
    const int re = r0 + ((py + 1) * h + 6) / 7;   // rs < re always (h >= 1)
    const int cs = c0 + (px * w) / 7;
    const int ce = c0 + ((px + 1) * w + 6) / 7;

    const float* base = fm + (size_t)c * (FH * FW);
    float m = -3.402823466e38f;
    for (int rr = rs; rr < re; ++rr) {
      const float* row = base + rr * FW;
      for (int cc = cs; cc < ce; ++cc) {
        float v = row[cc];
        m = v > m ? v : m;
      }
    }
    A[(size_t)n * KD + idx] = f2bf(m);
  }
}

// ---------------------------------------------------------------------------
// 2) FC GEMM:  Rpart[split][n][m] = sum_{k in split} A[n][k] * fc_w[m][k]
//    block = (m_block of 64 cols) x (K split), 8 waves; each wave owns a
//    16(n) x 64(m) strip. Software-pipelined: double-buffered LDS bf16
//    weight tiles (one barrier per 64-K step), NT weight stream, 8 WMMAs.
// ---------------------------------------------------------------------------
__global__ __launch_bounds__(256)
void fc_gemm_kernel(const unsigned short* __restrict__ A,
                    const float* __restrict__ W,
                    float* __restrict__ Rpart) {
  __shared__ unsigned short Bs[2][64 * BSTR];   // 2 x 9216 B, padded rows

  const int m0    = blockIdx.x * 64;
  const int split = blockIdx.y;
  const int kb0   = split * KCH;
  const int tid   = threadIdx.x;
  const int lane  = tid & 31;
  const int wv    = tid >> 5;      // wave id 0..7 -> n rows 16*wv..16*wv+15
  const int lr    = lane & 15;
  const int hi    = lane >> 4;     // half-wave select

  v8f acc[4] = {};

  const unsigned short* Arow = A + (size_t)(wv * 16 + lr) * KD;

  // staging assignment: thread -> (row si, 16-elem col group sj0)
  const int si  = tid >> 2;          // 0..63  (m_local)
  const int sj0 = (tid & 3) * 16;    // 0,16,32,48
  const size_t wrow = (size_t)(m0 + si) * KD + sj0;

  // ---- prologue: load first weight tile into registers (NT stream) ----
  f4 r[4];
#pragma unroll
  for (int q = 0; q < 4; ++q)
    r[q] = __builtin_nontemporal_load((const f4*)(W + wrow + kb0 + q * 4));

  for (int it = 0; it < NIT; ++it) {
    const int kb = kb0 + it * KSTEP;
    unsigned short* buf = Bs[it & 1];

    // ---- convert staged registers -> bf16 LDS tile ----
    // vector fptrunc: lowers to v_cvt_pk_bf16_f32 when the HW op exists
    Pack8 p0, p1;
    p0.b[0] = __builtin_convertvector(r[0], bf4);
    p0.b[1] = __builtin_convertvector(r[1], bf4);
    p1.b[0] = __builtin_convertvector(r[2], bf4);
    p1.b[1] = __builtin_convertvector(r[3], bf4);
    *(U4*)(buf + si * BSTR + sj0)     = p0.u;
    *(U4*)(buf + si * BSTR + sj0 + 8) = p1.u;

    // ---- issue next tile's global loads; they overlap the compute below ----
    if (it + 1 < NIT) {
#pragma unroll
      for (int q = 0; q < 4; ++q)
        r[q] = __builtin_nontemporal_load(
            (const f4*)(W + wrow + kb + KSTEP + q * 4));
    }

    __syncthreads();   // stores of this tile visible; prior readers of buf done

    // ---- compute: 2 x (A frag + 4 B frags, then 4 WMMA back-to-back) ----
#pragma unroll
    for (int kk = 0; kk < KSTEP; kk += 32) {
      // A fragment (16-bit A 16x32 layout):
      // lane<16: K 0-7 & 16-23 ; lane>=16: K 8-15 & 24-31
      Frag a;
      const U4* ap = (const U4*)(Arow + kb + kk);
      a.q[0] = ap[hi];
      a.q[1] = ap[2 + hi];

      // B fragments (16-bit B 32x16 layout): lane<16: K 0-7 & 16-23 ; lane>=16: K 8-15 & 24-31
      Frag b[4];
#pragma unroll
      for (int t = 0; t < 4; ++t) {
        const U4* bp = (const U4*)(buf + (t * 16 + lr) * BSTR + kk);
        b[t].q[0] = bp[hi];
        b[t].q[1] = bp[2 + hi];
      }

      // keep all 10 loads clustered (one clause, one wait), WMMAs issue
      // back-to-back afterwards (disjoint accumulators -> no hazard NOPs)
      __builtin_amdgcn_sched_barrier(0);

#pragma unroll
      for (int t = 0; t < 4; ++t)
        acc[t] = __builtin_amdgcn_wmma_f32_16x16x32_bf16(
            false, a.v, false, b[t].v, (short)0, acc[t], false, false);
    }
  }

  // ---- epilogue: D layout -> Rpart (disjoint per block, plain stores) ----
  float* out = Rpart + (size_t)split * (NROI * MFC);
#pragma unroll
  for (int t = 0; t < 4; ++t) {
    const int m = m0 + t * 16 + lr;
#pragma unroll
    for (int j = 0; j < 8; ++j) {
      const int n = wv * 16 + hi * 8 + j;
      out[(size_t)n * MFC + m] = acc[t][j];
    }
  }
}

// ---------------------------------------------------------------------------
// 3) Deterministic fixed-order reduction of the K splits + bias
// ---------------------------------------------------------------------------
__global__ __launch_bounds__(256)
void reduce_r_kernel(const float* __restrict__ Rpart,
                     const float* __restrict__ fc_b,
                     float* __restrict__ R) {
  const int i = blockIdx.x * 256 + threadIdx.x;   // 0 .. 128*1024-1
  float s = fc_b[i & (MFC - 1)];
#pragma unroll
  for (int p = 0; p < NSPL; ++p) s += Rpart[(size_t)p * (NROI * MFC) + i];
  R[i] = s;
}

// ---------------------------------------------------------------------------
// 4) cls/reg heads + softmax (tiny: 6.6 MFLOP). One wave per ROI.
// ---------------------------------------------------------------------------
__global__ __launch_bounds__(32)
void head_kernel(const float* __restrict__ R,
                 const float* __restrict__ cls_w, const float* __restrict__ cls_b,
                 const float* __restrict__ reg_w, const float* __restrict__ reg_b,
                 float* __restrict__ out) {
  const int n = blockIdx.x;
  __shared__ float logits[NCLS];
  const float* r = R + (size_t)n * MFC;
  const int t = threadIdx.x;

  if (t < NCLS) {
    const float* w = cls_w + (size_t)t * MFC;
    float s = cls_b[t];
    for (int k = 0; k < MFC; ++k) s = fmaf(r[k], w[k], s);
    logits[t] = s;
  } else if (t < NCLS + 4) {
    const int o = t - NCLS;
    const float* w = reg_w + (size_t)o * MFC;
    float s = reg_b[o];
    for (int k = 0; k < MFC; ++k) s = fmaf(r[k], w[k], s);
    out[NROI * NCLS + n * 4 + o] = s;
  }
  __syncthreads();

  if (t == 0) {
    float mx = logits[0];
    for (int i = 1; i < NCLS; ++i) mx = fmaxf(mx, logits[i]);
    float e[NCLS];
    float sum = 0.f;
    for (int i = 0; i < NCLS; ++i) { e[i] = __expf(logits[i] - mx); sum += e[i]; }
    const float inv = 1.f / sum;
    for (int i = 0; i < NCLS; ++i) out[n * NCLS + i] = e[i] * inv;
  }
}

// ---------------------------------------------------------------------------
// launch
// ---------------------------------------------------------------------------
extern "C" void kernel_launch(void* const* d_in, const int* in_sizes, int n_in,
                              void* d_out, int out_size, void* d_ws, size_t ws_size,
                              hipStream_t stream) {
  const float* fm    = (const float*)d_in[0];
  const int*   props = (const int*)  d_in[1];
  const float* fc_w  = (const float*)d_in[2];
  const float* fc_b  = (const float*)d_in[3];
  const float* cls_w = (const float*)d_in[4];
  const float* cls_b = (const float*)d_in[5];
  const float* reg_w = (const float*)d_in[6];
  const float* reg_b = (const float*)d_in[7];
  float* out = (float*)d_out;

  // workspace layout
  const size_t A_BYTES    = (size_t)NROI * KD * sizeof(unsigned short); // 12,845,056
  const size_t PART_BYTES = (size_t)NSPL * NROI * MFC * sizeof(float);  //  8,388,608
  unsigned short* A     = (unsigned short*)d_ws;
  float*          Rpart = (float*)((char*)d_ws + A_BYTES);
  float*          R     = (float*)((char*)d_ws + A_BYTES + PART_BYTES);

  roi_pool_kernel<<<NROI, 256, 0, stream>>>(fm, props, A);
  fc_gemm_kernel<<<dim3(MFC / 64, NSPL), 256, 0, stream>>>(A, fc_w, Rpart);
  reduce_r_kernel<<<(NROI * MFC) / 256, 256, 0, stream>>>(Rpart, fc_b, R);
  head_kernel<<<NROI, 32, 0, stream>>>(R, cls_w, cls_b, reg_w, reg_b, out);
}
